// SFLLayer_65317862638133
// MI455X (gfx1250) — compile-verified, matrix operand order checked by reading, not run on hardware
//
#include <hip/hip_runtime.h>
#include <hip/hip_bf16.h>
#include <math.h>

typedef __attribute__((ext_vector_type(16))) __bf16 v16bf;
typedef __attribute__((ext_vector_type(8)))  float  v8f;
typedef __attribute__((ext_vector_type(4)))  int    v4i;

#define B_    4
#define C_    256
#define H_    256
#define W_    256
#define HH_   32
#define WW_   32
#define NTOK_ 1024   // HH_*WW_
#define P_    64     // pixels per super-token (8x8)

#if defined(__AMDGCN__) && __has_builtin(__builtin_amdgcn_global_load_async_to_lds_b128)
#define HAVE_ASYNC_COPY 1
typedef __attribute__((address_space(1))) v4i as1_v4i;
typedef __attribute__((address_space(3))) v4i as3_v4i;
#else
#define HAVE_ASYNC_COPY 0
#endif

// float -> bf16 (round-to-nearest-even), no reliance on __bf16 arithmetic
static __device__ __forceinline__ __bf16 f2bf(float f) {
  unsigned u = __builtin_bit_cast(unsigned, f);
  unsigned r = (u + 0x7FFFu + ((u >> 16) & 1u)) >> 16;
  unsigned short s = (unsigned short)r;
  __bf16 o;
  __builtin_memcpy(&o, &s, 2);
  return o;
}

// 16-bit A/B WMMA fragment K-index map (CDNA5 ISA 7.12.2):
// lanes 0-15 hold K {0..7,16..23}, lanes 16-31 hold K {8..15,24..31}
static __device__ __forceinline__ int kmap16(int e, int lane) {
  int half = (lane >= 16) ? 8 : 0;
  return (e < 8) ? (half + e) : (16 + half + e - 8);
}

// ---------------------------------------------------------------------------
// Kernel A1: x2 = x + dwconv3x3(x) + b   (writes the residual base into d_out)
// ---------------------------------------------------------------------------
__global__ void k_dwc3_res(const float* __restrict__ x, const float* __restrict__ w,
                           const float* __restrict__ b, float* __restrict__ out) {
  int y  = blockIdx.x & (H_ - 1);
  int c  = (blockIdx.x >> 8) & (C_ - 1);
  int bb = blockIdx.x >> 16;
  int xx = threadIdx.x;
  const float* base = x + ((size_t)bb * C_ + c) * H_ * W_;
  float acc = b[c];
#pragma unroll
  for (int u = 0; u < 3; ++u) {
    int yy = y + u - 1;
    if (yy < 0 || yy >= H_) continue;
#pragma unroll
    for (int v = 0; v < 3; ++v) {
      int xv = xx + v - 1;
      if (xv < 0 || xv >= W_) continue;
      acc += w[c * 9 + u * 3 + v] * base[(size_t)yy * W_ + xv];
    }
  }
  size_t idx = (((size_t)bb * C_ + c) * H_ + y) * W_ + xx;
  out[idx] = x[idx] + acc;
}

// ---------------------------------------------------------------------------
// Kernel A2: channel LayerNorm per pixel -> xln (bf16); 8x8 block sums -> stok
// ---------------------------------------------------------------------------
__global__ void k_ln_stok(const float* __restrict__ x2, const float* __restrict__ lw,
                          const float* __restrict__ lb, __bf16* __restrict__ xln,
                          float* __restrict__ stok) {
  int y  = blockIdx.x & (H_ - 1);
  int bb = blockIdx.x >> 8;
  int xx = threadIdx.x;
  size_t base = (size_t)bb * C_ * H_ * W_ + (size_t)y * W_ + xx;
  float s = 0.f, q = 0.f;
  for (int c = 0; c < C_; ++c) {
    float v = x2[base + (size_t)c * H_ * W_];
    s += v; q += v * v;
  }
  float mu  = s * (1.0f / C_);
  float var = q * (1.0f / C_) - mu * mu;
  float rs  = rsqrtf(var + 1e-6f);
  int gy = y >> 3, gx = xx >> 3;
  for (int c = 0; c < C_; ++c) {
    float v  = x2[base + (size_t)c * H_ * W_];
    float xn = (v - mu) * rs * lw[c] + lb[c];
    xln[base + (size_t)c * H_ * W_] = f2bf(xn);
    float g = xn;
    g += __shfl_xor(g, 1, 8);
    g += __shfl_xor(g, 2, 8);
    g += __shfl_xor(g, 4, 8);
    if ((xx & 7) == 0)
      atomicAdd(&stok[(((size_t)bb * C_ + c) * HH_ + gy) * WW_ + gx], g);
  }
}

// ---------------------------------------------------------------------------
// Kernel B: per super-token attention iteration.
//  pf staged global->LDS via GLOBAL_LOAD_ASYNC_TO_LDS_B128 ([c][p] copy),
//  then transposed in LDS to [p][c] so BOTH GEMMs get contiguous A operands.
//  GEMM1 (WMMA): logits(64x16) = pf(64x256) @ patchesT(256x16)
//  softmax over 9; GEMM2 (WMMA): st(256x9) = pf^T(256x64) @ aff(64x16)
// ---------------------------------------------------------------------------
#define PFS_ 72   // padded pixel stride (keeps 16B alignment of 8-pixel rows)

__global__ void __launch_bounds__(128)
k_token_iter(const __bf16* __restrict__ xln, const float* __restrict__ stok,
             float sscale, float* __restrict__ affg, float* __restrict__ asumg,
             float* __restrict__ stg) {
  __shared__ __bf16 pf_cp[C_][PFS_];    // [c][p]: async-staged, GEMM2 A
  __shared__ __bf16 pf_pc[P_][C_ + 8];  // [p][c]: LDS-transposed, GEMM1 A
  __shared__ __bf16 pat[16][C_];        // pat[col][c], cols 9..15 zero
  __shared__ float  logit[P_][16];
  __shared__ __bf16 affbT[16][P_];      // aff^T[col][p], GEMM2 B

  int n  = blockIdx.x & (NTOK_ - 1);
  int bb = blockIdx.x >> 10;
  int ti = n >> 5, tj = n & 31;
  int tid  = threadIdx.x;
  int wave = tid >> 5, lane = tid & 31;

  // ---- stage pf: 2048 chunks of 8 bf16 (16B), global-contiguous per (c,row)
#if HAVE_ASYNC_COPY
#pragma unroll
  for (int i = 0; i < 16; ++i) {
    int idx = tid + i * 128;          // 0..2047
    int c  = idx >> 3;
    int py = idx & 7;
    const __bf16* g = xln + (((size_t)bb * C_ + c) * H_ + (ti * 8 + py)) * W_ + tj * 8;
    __builtin_amdgcn_global_load_async_to_lds_b128((as1_v4i*)g,
                                                   (as3_v4i*)&pf_cp[c][py * 8], 0, 0);
  }
#else
  for (int idx = tid; idx < C_ * P_; idx += 128) {
    int c = idx >> 6;
    int p = idx & 63;
    pf_cp[c][p] = xln[(((size_t)bb * C_ + c) * H_ + (ti * 8 + (p >> 3))) * W_ + tj * 8 + (p & 7)];
  }
#endif
  // patch gather overlaps the async DMA
  for (int idx = tid; idx < 16 * C_; idx += 128) {
    int c   = idx >> 4;
    int col = idx & 15;
    float v = 0.f;
    if (col < 9) {
      int si = ti + (col / 3) - 1;
      int sj = tj + (col % 3) - 1;
      if (si >= 0 && si < HH_ && sj >= 0 && sj < WW_)
        v = stok[(((size_t)bb * C_ + c) * HH_ + si) * WW_ + sj];
    }
    pat[col][c] = f2bf(v);
  }
#if HAVE_ASYNC_COPY
  asm volatile("s_wait_asynccnt 0" ::: "memory");
#endif
  __syncthreads();

  // ---- LDS transpose: pf_pc[p][c] = pf_cp[c][p] (b128 reads, u16 scatters)
#pragma unroll
  for (int i = 0; i < 16; ++i) {
    int idx = tid + i * 128;          // 0..2047
    int c  = idx >> 3;
    int pb = idx & 7;
    union { v4i v; __bf16 e[8]; } t;
    t.v = *(const v4i*)&pf_cp[c][pb * 8];
#pragma unroll
    for (int q = 0; q < 8; ++q) pf_pc[pb * 8 + q][c] = t.e[q];
  }
  __syncthreads();

  // GEMM1: each wave owns one 16-row M tile of pf (A and B both contiguous)
  {
    union { v8f v; float e[8]; } acc;
#pragma unroll
    for (int r = 0; r < 8; ++r) acc.e[r] = 0.f;
    int m    = wave * 16 + (lane & 15);   // pixel p
    int ncol = lane & 15;
#pragma unroll
    for (int kk = 0; kk < 8; ++kk) {
      union { v16bf v; __bf16 e[16]; } a, bf;
#pragma unroll
      for (int e = 0; e < 16; ++e) {
        int k  = kk * 32 + kmap16(e, lane);   // channel
        a.e[e]  = pf_pc[m][k];
        bf.e[e] = pat[ncol][k];
      }
      acc.v = __builtin_amdgcn_wmma_f32_16x16x32_bf16(false, a.v, false, bf.v,
                                                      (short)0, acc.v, false, false);
    }
#pragma unroll
    for (int r = 0; r < 8; ++r) {
      int row = wave * 16 + r + ((lane >= 16) ? 8 : 0);
      logit[row][lane & 15] = acc.e[r];
    }
  }
  __syncthreads();

  // softmax over 9 (C^-0.5 = 1/16; sscale folds the stoken mean scaling)
  if (tid < P_) {
    const float scale = 0.0625f * sscale;
    float l[9];
    float mx = -1e30f;
#pragma unroll
    for (int k = 0; k < 9; ++k) { l[k] = logit[tid][k] * scale; mx = fmaxf(mx, l[k]); }
    float ssum = 0.f;
#pragma unroll
    for (int k = 0; k < 9; ++k) { l[k] = __expf(l[k] - mx); ssum += l[k]; }
    float inv = 1.0f / ssum;
    size_t gbase = (((size_t)bb * NTOK_ + n) * P_ + tid) * 16;
#pragma unroll
    for (int k = 0; k < 16; ++k) {
      float av = (k < 9) ? l[k] * inv : 0.f;
      logit[tid][k]  = av;          // reuse as f32 aff
      affbT[k][tid]  = f2bf(av);    // transposed for contiguous B fragments
      affg[gbase + k] = av;
    }
  }
  __syncthreads();

  if (tid < 9) {
    float s = 0.f;
    for (int r = 0; r < P_; ++r) s += logit[r][tid];
    asumg[((size_t)bb * NTOK_ + n) * 16 + tid] = s;
  }

  // GEMM2: st = pf^T @ aff ; A = pf_cp[c][p], B = affbT[col][p] (both contiguous)
  for (int mt = wave; mt < 16; mt += 4) {
    union { v8f v; float e[8]; } acc;
#pragma unroll
    for (int r = 0; r < 8; ++r) acc.e[r] = 0.f;
    int m    = mt * 16 + (lane & 15);   // channel
    int ncol = lane & 15;               // aff column
#pragma unroll
    for (int kk = 0; kk < 2; ++kk) {
      union { v16bf v; __bf16 e[16]; } a, bf;
#pragma unroll
      for (int e = 0; e < 16; ++e) {
        int k  = kk * 32 + kmap16(e, lane);   // pixel p
        a.e[e]  = pf_cp[m][k];
        bf.e[e] = affbT[ncol][k];
      }
      acc.v = __builtin_amdgcn_wmma_f32_16x16x32_bf16(false, a.v, false, bf.v,
                                                      (short)0, acc.v, false, false);
    }
    if (ncol < 9) {
#pragma unroll
      for (int r = 0; r < 8; ++r) {
        int c = mt * 16 + r + ((lane >= 16) ? 8 : 0);
        stg[(((size_t)bb * NTOK_ + n) * C_ + c) * 9 + ncol] = acc.e[r];
      }
    }
  }
}

// ---------------------------------------------------------------------------
// Kernel C: fold 3x3 neighborhood of st and normalize by folded asum
// ---------------------------------------------------------------------------
__global__ void k_fold(const float* __restrict__ stg, const float* __restrict__ asumg,
                       float* __restrict__ stok_out) {
  __shared__ float asf;
  int n  = blockIdx.x & (NTOK_ - 1);
  int bb = blockIdx.x >> 10;
  int i = n >> 5, j = n & 31;
  int c = threadIdx.x;
  if (c == 0) {
    float s = 0.f;
    for (int ki = 0; ki < 3; ++ki)
      for (int kj = 0; kj < 3; ++kj) {
        int i2 = i - (ki - 1), j2 = j - (kj - 1);
        if (i2 >= 0 && i2 < HH_ && j2 >= 0 && j2 < WW_)
          s += asumg[((size_t)bb * NTOK_ + i2 * WW_ + j2) * 16 + ki * 3 + kj];
      }
    asf = s + 1e-12f;
  }
  __syncthreads();
  float s = 0.f;
  for (int ki = 0; ki < 3; ++ki)
    for (int kj = 0; kj < 3; ++kj) {
      int i2 = i - (ki - 1), j2 = j - (kj - 1);
      if (i2 >= 0 && i2 < HH_ && j2 >= 0 && j2 < WW_)
        s += stg[(((size_t)bb * NTOK_ + i2 * WW_ + j2) * C_ + c) * 9 + ki * 3 + kj];
    }
  stok_out[(((size_t)bb * C_ + c) * HH_ + i) * WW_ + j] = s / asf;
}

// ---------------------------------------------------------------------------
// Kernel D: conv1x1 as WMMA GEMM: out[b,o,p] = sum_c W[o,c]*x[b,c,p] + bias[o]
// ---------------------------------------------------------------------------
__global__ void __launch_bounds__(128)
k_conv1x1(const float* __restrict__ xin, const float* __restrict__ wmat,
          const float* __restrict__ bias, float* __restrict__ out) {
  int ng = blockIdx.x & 15;
  int mt = (blockIdx.x >> 4) & 15;
  int bb = blockIdx.x >> 8;
  int wave = threadIdx.x >> 5, lane = threadIdx.x & 31;
  int nt   = ng * 4 + wave;
  int m    = mt * 16 + (lane & 15);
  int ncol = nt * 16 + (lane & 15);
  union { v8f v; float e[8]; } acc;
#pragma unroll
  for (int r = 0; r < 8; ++r) acc.e[r] = 0.f;
  for (int kk = 0; kk < 8; ++kk) {
    union { v16bf v; __bf16 e[16]; } a, bf;
#pragma unroll
    for (int e = 0; e < 16; ++e) {
      int k  = kk * 32 + kmap16(e, lane);
      a.e[e]  = f2bf(wmat[(size_t)m * C_ + k]);
      bf.e[e] = f2bf(xin[((size_t)bb * C_ + k) * NTOK_ + ncol]);
    }
    acc.v = __builtin_amdgcn_wmma_f32_16x16x32_bf16(false, a.v, false, bf.v,
                                                    (short)0, acc.v, false, false);
  }
#pragma unroll
  for (int r = 0; r < 8; ++r) {
    int mo = mt * 16 + r + ((lane >= 16) ? 8 : 0);
    out[((size_t)bb * C_ + mo) * NTOK_ + ncol] = acc.e[r] + bias[mo];
  }
}

// ---------------------------------------------------------------------------
// Kernel E: depthwise 5x5 (pad=2) on 32x32 token map
// ---------------------------------------------------------------------------
__global__ void k_dwc5(const float* __restrict__ xin, const float* __restrict__ w,
                       const float* __restrict__ bias, float* __restrict__ out) {
  int c  = blockIdx.x & 255;
  int bb = blockIdx.x >> 8;
  const float* base = xin + ((size_t)bb * C_ + c) * NTOK_;
  float wl[25];
#pragma unroll
  for (int t = 0; t < 25; ++t) wl[t] = w[c * 25 + t];
  for (int p = threadIdx.x; p < NTOK_; p += 256) {
    int i = p >> 5, j = p & 31;
    float acc = bias[c];
#pragma unroll
    for (int u = 0; u < 5; ++u) {
      int ii = i + u - 2;
      if (ii < 0 || ii >= HH_) continue;
#pragma unroll
      for (int v = 0; v < 5; ++v) {
        int jj = j + v - 2;
        if (jj < 0 || jj >= WW_) continue;
        acc += wl[u * 5 + v] * base[ii * WW_ + jj];
      }
    }
    out[((size_t)bb * C_ + c) * NTOK_ + p] = acc;
  }
}

// ---------------------------------------------------------------------------
// Kernel G: po = sp(256x9pad32) @ aff^T(9pad32 x 64) ; out += gamma * po
// ---------------------------------------------------------------------------
__global__ void __launch_bounds__(128)
k_project_out(const float* __restrict__ stokf, const float* __restrict__ affg,
              const float* __restrict__ gamma, float* __restrict__ out) {
  __shared__ __bf16 spA[C_][32];   // A[m=c][k]
  __shared__ __bf16 afb[P_][32];   // B[k][n=p] stored as afb[p][k]
  int n  = blockIdx.x & (NTOK_ - 1);
  int bb = blockIdx.x >> 10;
  int ti = n >> 5, tj = n & 31;
  int tid  = threadIdx.x;
  int wave = tid >> 5, lane = tid & 31;

  for (int idx = tid; idx < C_ * 32; idx += 128) spA[idx >> 5][idx & 31] = f2bf(0.f);
  for (int idx = tid; idx < P_ * 32; idx += 128) afb[idx >> 5][idx & 31] = f2bf(0.f);
  __syncthreads();
  for (int idx = tid; idx < C_ * 9; idx += 128) {
    int c = idx / 9, k = idx % 9;
    int si = ti + k / 3 - 1, sj = tj + k % 3 - 1;
    float v = 0.f;
    if (si >= 0 && si < HH_ && sj >= 0 && sj < WW_)
      v = stokf[(((size_t)bb * C_ + c) * HH_ + si) * WW_ + sj];
    spA[c][k] = f2bf(v);
  }
  for (int idx = tid; idx < P_ * 9; idx += 128) {
    int p = idx / 9, k = idx % 9;
    afb[p][k] = f2bf(affg[(((size_t)bb * NTOK_ + n) * P_ + p) * 16 + k]);
  }
  __syncthreads();

  for (int t = wave; t < 64; t += 4) {
    int mt = t >> 2, nt = t & 3;
    int m  = mt * 16 + (lane & 15);
    int pc = nt * 16 + (lane & 15);
    union { v8f v; float e[8]; } acc;
#pragma unroll
    for (int r = 0; r < 8; ++r) acc.e[r] = 0.f;
    union { v16bf v; __bf16 e[16]; } a, bf;
#pragma unroll
    for (int e = 0; e < 16; ++e) {
      int k = kmap16(e, lane);
      a.e[e]  = spA[m][k];
      bf.e[e] = afb[pc][k];
    }
    acc.v = __builtin_amdgcn_wmma_f32_16x16x32_bf16(false, a.v, false, bf.v,
                                                    (short)0, acc.v, false, false);
#pragma unroll
    for (int r = 0; r < 8; ++r) {
      int c = mt * 16 + r + ((lane >= 16) ? 8 : 0);
      int y = ti * 8 + (pc >> 3), x = tj * 8 + (pc & 7);
      size_t oidx = (((size_t)bb * C_ + c) * H_ + y) * W_ + x;
      out[oidx] += gamma[c] * acc.e[r];
    }
  }
}

// ---------------------------------------------------------------------------
extern "C" void kernel_launch(void* const* d_in, const int* in_sizes, int n_in,
                              void* d_out, int out_size, void* d_ws, size_t ws_size,
                              hipStream_t stream) {
  const float* x     = (const float*)d_in[0];
  const float* dwc_w = (const float*)d_in[1];
  const float* dwc_b = (const float*)d_in[2];
  const float* ln_w  = (const float*)d_in[3];
  const float* ln_b  = (const float*)d_in[4];
  const float* r1_w  = (const float*)d_in[5];
  const float* r1_b  = (const float*)d_in[6];
  const float* rd_w  = (const float*)d_in[7];
  const float* rd_b  = (const float*)d_in[8];
  const float* r2_w  = (const float*)d_in[9];
  const float* r2_b  = (const float*)d_in[10];
  const float* gamma = (const float*)d_in[11];
  float* out = (float*)d_out;

  char* ws = (char*)d_ws;
  size_t off = 0;
  __bf16* xln  = (__bf16*)(ws + off); off += (size_t)B_ * C_ * H_ * W_ * 2;
  float* stok0 = (float*)(ws + off);  off += (size_t)B_ * C_ * HH_ * WW_ * 4;
  float* stok1 = (float*)(ws + off);  off += (size_t)B_ * C_ * HH_ * WW_ * 4;
  float* stok2 = (float*)(ws + off);  off += (size_t)B_ * C_ * HH_ * WW_ * 4;
  float* r1o   = (float*)(ws + off);  off += (size_t)B_ * C_ * HH_ * WW_ * 4;
  float* dwo   = (float*)(ws + off);  off += (size_t)B_ * C_ * HH_ * WW_ * 4;
  float* r2o   = (float*)(ws + off);  off += (size_t)B_ * C_ * HH_ * WW_ * 4;
  float* affg  = (float*)(ws + off);  off += (size_t)B_ * NTOK_ * P_ * 16 * 4;
  float* asumg = (float*)(ws + off);  off += (size_t)B_ * NTOK_ * 16 * 4;
  float* stg   = (float*)(ws + off);  off += (size_t)B_ * NTOK_ * C_ * 9 * 4;

  (void)hipMemsetAsync(stok0, 0, (size_t)B_ * C_ * HH_ * WW_ * 4, stream);
  k_dwc3_res<<<B_ * C_ * H_, W_, 0, stream>>>(x, dwc_w, dwc_b, out);
  k_ln_stok<<<B_ * H_, W_, 0, stream>>>(out, ln_w, ln_b, xln, stok0);
  k_token_iter<<<B_ * NTOK_, 128, 0, stream>>>(xln, stok0, 1.0f / 64.0f, affg, asumg, stg);
  k_fold<<<B_ * NTOK_, C_, 0, stream>>>(stg, asumg, stok1);
  k_token_iter<<<B_ * NTOK_, 128, 0, stream>>>(xln, stok1, 1.0f, affg, asumg, stg);
  k_fold<<<B_ * NTOK_, C_, 0, stream>>>(stg, asumg, stok2);
  k_conv1x1<<<B_ * 16 * 16, 128, 0, stream>>>(stok2, r1_w, r1_b, r1o);
  k_dwc5<<<B_ * C_, 256, 0, stream>>>(r1o, rd_w, rd_b, dwo);
  k_conv1x1<<<B_ * 16 * 16, 128, 0, stream>>>(dwo, r2_w, r2_b, r2o);
  k_project_out<<<B_ * NTOK_, 128, 0, stream>>>(r2o, affg, gamma, out);

  (void)in_sizes; (void)n_in; (void)out_size; (void)ws_size;
}
